// VectorQuantizer_71640054497909
// MI455X (gfx1250) — matrix operand order repaired and checked by reading, not compiled, not run on hardware
//
#include <hip/hip_runtime.h>

#define N_TOK   16384
#define DIM     256
#define K_CODES 8192
#define TILE_M  16
#define TILE_N  16
#define NEG_INF (-3.402823466e38f)

// LDS panel: 64 codebook rows of 512B, padded to 528B stride by the TDM pad
// feature (132 dwords == 4 mod 64 -> conflict-free bank spread for 16 lanes).
#define PANEL_CODES   64
#define ROW_BYTES     512
#define LDS_ROW_BYTES 528
#define PANEL_BYTES   (PANEL_CODES * LDS_ROW_BYTES)   // 33792
#define N_PANELS      (K_CODES / PANEL_CODES)         // 128

typedef __attribute__((ext_vector_type(16))) _Float16     v16h;
typedef __attribute__((ext_vector_type(8)))  _Float16     v8h;
typedef __attribute__((ext_vector_type(8)))  float        v8f;
typedef __attribute__((ext_vector_type(4)))  unsigned int u32x4;
typedef __attribute__((ext_vector_type(8)))  int          i32x8;
typedef __attribute__((ext_vector_type(4)))  int          i32x4;

#if defined(__has_builtin)
#if __has_builtin(__builtin_amdgcn_tensor_load_to_lds) && \
    __has_builtin(__builtin_amdgcn_s_wait_tensorcnt)
#define HAVE_TDM 1
#endif
#endif
#ifndef HAVE_TDM
#define HAVE_TDM 0
#endif

// -------- Kernel 1: L2-normalize rows (fp32 -> f16), one wave32 per row ----
__global__ void vq_normalize_rows(const float* __restrict__ in,
                                  _Float16* __restrict__ out, int rows) {
  int wave = (int)((blockIdx.x * blockDim.x + threadIdx.x) >> 5);
  int lane = threadIdx.x & 31;
  if (wave >= rows) return;
  const float* r = in + (size_t)wave * DIM + lane * 8;
  float vals[8];
  float sq = 0.f;
#pragma unroll
  for (int i = 0; i < 8; ++i) {
    vals[i] = r[i];
    sq += vals[i] * vals[i];
  }
#pragma unroll
  for (int m = 16; m >= 1; m >>= 1) sq += __shfl_xor(sq, m, 32);
  float inv = rsqrtf(fmaxf(sq, 1e-12f));
  _Float16* o = out + (size_t)wave * DIM + lane * 8;
#pragma unroll
  for (int i = 0; i < 8; ++i) o[i] = (_Float16)(vals[i] * inv);
}

// -------- TDM panel load: 2D tile 64 rows x 512B, LDS rows padded to 528B --
#if HAVE_TDM
__device__ __forceinline__ void tdm_load_panel(const _Float16* gsrc,
                                               unsigned int ldsOff) {
  unsigned long long ga = (unsigned long long)(uintptr_t)gsrc;
  u32x4 g0;
  g0[0] = 1u;                                     // count=1, user descriptor
  g0[1] = ldsOff;                                 // lds_addr
  g0[2] = (unsigned int)ga;                       // global_addr[31:0]
  g0[3] = (unsigned int)(ga >> 32) | 0x80000000u; // global_addr[56:32] | type=2
  i32x8 g1;
  // data_size=3 (8B units) | pad_enable | pad_interval=6 (128 dw = 512B)
  // | pad_amount=3 (4 dw = 16B)  -> LDS row stride 528B
  g1[0] = (3 << 16) | (1 << 20) | (6 << 22) | (3 << 25);
  g1[1] = (int)(64u << 16);       // tensor_dim0 = 64 units (512B row)
  g1[2] = (int)(8192u << 16);     // tensor_dim1 = 8192 rows
  g1[3] = (int)(64u << 16);       // tile_dim0 = 64 units
  g1[4] = PANEL_CODES;            // tile_dim1 = 64 rows, tile_dim2 = 0
  g1[5] = 64;                     // tensor_dim0_stride = 64 units (lo32)
  g1[6] = (int)(64u << 16);       // d0_stride hi=0 | tensor_dim1_stride lo16
  g1[7] = 0;
  i32x4 z4 = {0, 0, 0, 0};
#if __clang_major__ >= 23
  i32x8 z8 = {0, 0, 0, 0, 0, 0, 0, 0};
  __builtin_amdgcn_tensor_load_to_lds(g0, g1, z4, z4, z8, 0);
#else
  __builtin_amdgcn_tensor_load_to_lds(g0, g1, z4, z4, 0);
#endif
}
#endif

__device__ __forceinline__ void panel_issue(const _Float16* gsrc, char* ldsDst) {
#if HAVE_TDM
  if (threadIdx.x == 0)   // one TDM descriptor per workgroup (EXEC ignored)
    tdm_load_panel(gsrc, (unsigned int)(uintptr_t)(void*)ldsDst);
#else
  // cooperative synchronous copy with manual 16B row padding
  for (int i = threadIdx.x; i < PANEL_CODES * (ROW_BYTES / 16); i += 256) {
    int code = i >> 5, off = (i & 31) << 4;
    *(float4*)(ldsDst + code * LDS_ROW_BYTES + off) =
        *(const float4*)((const char*)gsrc + code * ROW_BYTES + off);
  }
#endif
}

__device__ __forceinline__ void panel_wait() {
#if HAVE_TDM
  if (threadIdx.x < 32) __builtin_amdgcn_s_wait_tensorcnt(0);
#endif
}

// -------- Kernel 2: fused f16 WMMA GEMM + per-row argmax, LDS-staged B ----
// Block = 8 waves; each wave owns 16 tokens; all waves share TDM-loaded
// codebook panels (double buffered). Inner loop interleaves two code tiles
// (independent accumulator chains) and prefetches the next k-chunk's B
// fragments so ds_load latency is hidden behind the WMMA chain.
__global__ void __launch_bounds__(256)
vq_argmax_wmma(const _Float16* __restrict__ fxn,
               const _Float16* __restrict__ cbn,
               int* __restrict__ indices) {
  extern __shared__ __align__(16) char smem[];  // 2 * PANEL_BYTES
  const int lane = threadIdx.x & 31;
  const int waveId = threadIdx.x >> 5;
  const int half = lane >> 4;           // 0: lanes 0-15, 1: lanes 16-31
  const int l15  = lane & 15;
  const int rowBase = (blockIdx.x * 8 + waveId) * TILE_M;

  // A fragments: 16x32 f16 per k-chunk, k = 16*half + s (same order as B), resident in VGPRs.
  v16h a[8];
  {
    const _Float16* arow = fxn + (size_t)(rowBase + l15) * DIM + (half ? 16 : 0);
#pragma unroll
    for (int c = 0; c < 8; ++c) {
      union { v16h v; v8h h[2]; } u;
      u.h[0] = *(const v8h*)(arow + c * 32);
      u.h[1] = *(const v8h*)(arow + c * 32 + 8);
      a[c] = u.v;
    }
  }

  float bestV[8];
  int   bestI[8];
#pragma unroll
  for (int r = 0; r < 8; ++r) { bestV[r] = NEG_INF; bestI[r] = 0; }

  // Prologue: panel 0 into buffer 0.
  panel_issue(cbn, smem);
  panel_wait();
  __syncthreads();

  for (int p = 0; p < N_PANELS; ++p) {
    // Overlap: TDM-load next panel while computing on current one.
    if (p + 1 < N_PANELS)
      panel_issue(cbn + (size_t)(p + 1) * PANEL_CODES * DIM,
                  smem + ((p + 1) & 1) * PANEL_BYTES);

    const char* base = smem + (p & 1) * PANEL_BYTES + half * 32;
#pragma unroll
    for (int tp = 0; tp < 2; ++tp) {   // two tile-pairs per 64-code panel
      const char* brow0 = base + ((tp * 2 + 0) * TILE_N + l15) * LDS_ROW_BYTES;
      const char* brow1 = base + ((tp * 2 + 1) * TILE_N + l15) * LDS_ROW_BYTES;
      v8f acc0 = {}, acc1 = {};
      v16h b0 = *(const v16h*)(brow0);
      v16h b1 = *(const v16h*)(brow1);
#pragma unroll
      for (int c = 0; c < 8; ++c) {
        v16h n0, n1;
        if (c < 7) {                    // prefetch next k-chunk before compute
          n0 = *(const v16h*)(brow0 + (c + 1) * 64);
          n1 = *(const v16h*)(brow1 + (c + 1) * 64);
        }
        acc0 = __builtin_amdgcn_wmma_f32_16x16x32_f16(
            false, a[c], false, b0, (short)0, acc0, false, false);
        acc1 = __builtin_amdgcn_wmma_f32_16x16x32_f16(
            false, a[c], false, b1, (short)0, acc1, false, false);
        if (c < 7) { b0 = n0; b1 = n1; }
      }
      const int col0 = p * PANEL_CODES + (tp * 2 + 0) * TILE_N + l15;
      const int col1 = p * PANEL_CODES + (tp * 2 + 1) * TILE_N + l15;
#pragma unroll
      for (int r = 0; r < 8; ++r) {
        float s0 = acc0[r];
        if (s0 > bestV[r]) { bestV[r] = s0; bestI[r] = col0; }
        float s1 = acc1[r];
        if (s1 > bestV[r]) { bestV[r] = s1; bestI[r] = col1; }
      }
    }

    panel_wait();      // next panel landed
    __syncthreads();   // all waves done reading the buffer we refill next
  }

  // Reduce the 16 columns held by each half-wave; tie -> smaller index.
#pragma unroll
  for (int r = 0; r < 8; ++r) {
    float v = bestV[r];
    int   i = bestI[r];
#pragma unroll
    for (int m = 8; m >= 1; m >>= 1) {
      float ov = __shfl_xor(v, m, 32);
      int   oi = __shfl_xor(i, m, 32);
      if (ov > v || (ov == v && oi < i)) { v = ov; i = oi; }
    }
    bestV[r] = v;
    bestI[r] = i;
  }
  if (l15 == 0) {
#pragma unroll
    for (int r = 0; r < 8; ++r)
      indices[rowBase + half * 8 + r] = bestI[r];
  }
}

// -------- Kernel 3: gather codebook rows, write quantized + idx, partial loss
__global__ void vq_gather_loss(const float* __restrict__ x,
                               const float* __restrict__ codebook,
                               const int* __restrict__ indices,
                               float* __restrict__ outQ,
                               float* __restrict__ outIdx,
                               float* __restrict__ partial) {
  __shared__ float s[256];
  const int t = blockIdx.x;
  const int d = threadIdx.x;
  const int idx = indices[t];
  float q  = codebook[(size_t)idx * DIM + d];
  float xv = x[(size_t)t * DIM + d];
  outQ[(size_t)t * DIM + d] = q;          // STE forward value == quantized
  float diff = q - xv;
  s[d] = diff * diff;
  __syncthreads();
  for (int stride = 128; stride >= 1; stride >>= 1) {
    if (d < stride) s[d] += s[d + stride];
    __syncthreads();
  }
  if (d == 0) {
    partial[t] = s[0];
    outIdx[t] = (float)idx;
  }
}

// -------- Kernel 4: deterministic final loss reduction ----
__global__ void vq_finalize(const float* __restrict__ partial,
                            float* __restrict__ lossOut) {
  __shared__ float s[256];
  float acc = 0.f;
  for (int i = threadIdx.x; i < N_TOK; i += 256) acc += partial[i];
  s[threadIdx.x] = acc;
  __syncthreads();
  for (int stride = 128; stride >= 1; stride >>= 1) {
    if (threadIdx.x < stride) s[threadIdx.x] += s[threadIdx.x + stride];
    __syncthreads();
  }
  if (threadIdx.x == 0)
    lossOut[0] = 1.25f * s[0] / ((float)N_TOK * (float)DIM);  // 0.25*c + 1*cb
}

extern "C" void kernel_launch(void* const* d_in, const int* in_sizes, int n_in,
                              void* d_out, int out_size, void* d_ws, size_t ws_size,
                              hipStream_t stream) {
  (void)in_sizes; (void)n_in; (void)out_size; (void)ws_size;
  const float* x        = (const float*)d_in[0];   // [16384, 256]
  const float* codebook = (const float*)d_in[1];   // [8192, 256]

  // Workspace carve-up (~12.2 MB total)
  char* ws = (char*)d_ws;
  _Float16* fxn = (_Float16*)ws;  ws += (size_t)N_TOK   * DIM * sizeof(_Float16);
  _Float16* cbn = (_Float16*)ws;  ws += (size_t)K_CODES * DIM * sizeof(_Float16);
  int*   indices = (int*)ws;      ws += (size_t)N_TOK * sizeof(int);
  float* partial = (float*)ws;    ws += (size_t)N_TOK * sizeof(float);

  // Output layout: quantized [16384*256] | loss [1] | indices-as-float [16384]
  float* outQ    = (float*)d_out;
  float* lossOut = outQ + (size_t)N_TOK * DIM;
  float* outIdx  = lossOut + 1;

  // 67.5 KB dynamic LDS (double-buffered panel) -- raise the cap (WGP: 320KB).
  hipFuncSetAttribute((const void*)vq_argmax_wmma,
                      hipFuncAttributeMaxDynamicSharedMemorySize,
                      2 * PANEL_BYTES);

  vq_normalize_rows<<<N_TOK   / 8, 256, 0, stream>>>(x,        fxn, N_TOK);
  vq_normalize_rows<<<K_CODES / 8, 256, 0, stream>>>(codebook, cbn, K_CODES);
  vq_argmax_wmma<<<(N_TOK / TILE_M) / 8, 256, 2 * PANEL_BYTES, stream>>>(fxn, cbn, indices);
  vq_gather_loss<<<N_TOK, 256, 0, stream>>>(x, codebook, indices, outQ, outIdx, partial);
  vq_finalize<<<1, 256, 0, stream>>>(partial, lossOut);
}
